// ESN_71674414235678
// MI455X (gfx1250) — matrix operand order repaired and checked by reading, not compiled, not run on hardware
//
#include <hip/hip_runtime.h>
#include <hip/hip_bf16.h>

// ---------------- problem constants ----------------
#define IN_DIM   128
#define HID      1024
#define TSTEPS   2048
#define BATCH    256
#define LASTN    20
#define OUT_DIM  32

// ---------------- tiling ----------------
#define MB        16          // batch rows per workgroup (WMMA M)
#define WAVES     8           // waves per workgroup
#define TPW       8           // 16-col N-tiles per wave  (8*8*16 = 1024 cols)
#define NTILES    (HID / 16)  // 64
#define HSTRIDE   (HID + 8)   // LDS bf16 row stride for h (bank-conflict pad)
#define XSTRIDE   (IN_DIM + 8)

typedef __attribute__((ext_vector_type(16))) __bf16 v16bf;
typedef __attribute__((ext_vector_type(8)))  __bf16 v8bf;
typedef __attribute__((ext_vector_type(8)))  float  v8f;

union ABfrag { v16bf v; struct { v8bf lo, hi; } p; };

__device__ __forceinline__ float fast_tanh(float x) {
#if __has_builtin(__builtin_amdgcn_tanhf)
    return __builtin_amdgcn_tanhf(x);   // native CDNA5 v_tanh_f32
#else
    return tanhf(x);
#endif
}

// -------------------------------------------------------------------------
// Pack a row-major [K,N] f32 matrix into bf16 WMMA B-fragments for
// v_wmma_f32_16x16x32_bf16.  Fragment (kc, nt): 32 lanes x 16 bf16.
//   lanes 0-15 : N = nt*16 + lane,      K = kc*32 + e      (e = 0..15)
//   lanes 16-31: N = nt*16 + lane-16,   K = kc*32 + 16 + e
// Flat element index = ((kc*NT + nt)*32 + lane)*16 + e.
// -------------------------------------------------------------------------
__global__ void pack_frags_kernel(const float* __restrict__ w,
                                  __bf16* __restrict__ frags,
                                  int K, int N) {
    int idx = blockIdx.x * 256 + threadIdx.x;       // one thread per (frag,lane)
    int total = (K >> 5) * (N >> 4) * 32;
    if (idx >= total) return;
    int lane = idx & 31;
    int frag = idx >> 5;
    int nt   = frag % (N >> 4);
    int kc   = frag / (N >> 4);
    int n    = nt * 16 + (lane & 15);
    int k0   = kc * 32 + ((lane >> 4) << 4);        // +16 for upper half-wave
    __bf16* dst = frags + (size_t)idx * 16;
#pragma unroll
    for (int e = 0; e < 16; ++e)
        dst[e] = (__bf16)w[(size_t)(k0 + e) * N + n];
}

// lin_w [32][20480] -> lin_wT [20480][32] for coalesced reads in the head.
__global__ void transpose_linw_kernel(const float* __restrict__ w,
                                      float* __restrict__ wt) {
    int idx = blockIdx.x * 256 + threadIdx.x;
    if (idx >= OUT_DIM * HID * LASTN) return;
    int k = idx >> 5;
    int o = idx & 31;
    wt[idx] = w[(size_t)o * (HID * LASTN) + k];
}

// -------------------------------------------------------------------------
// One GEMM phase: acc[M=16, wave's 128 cols] += A(bf16 LDS) * B(bf16 frags)
// A layout (matches packed B): lane<16 holds row=lane,
// K = kc*32 + {0..15}; lane>=16 holds row=lane-16, K = +16 offset.
// -------------------------------------------------------------------------
__device__ __forceinline__ void gemm_phase(const __bf16* src, int stride,
                                           int kchunks,
                                           const __bf16* __restrict__ frags,
                                           int ntbase, int lane, v8f* acc) {
    const int row   = lane & 15;
    const int khalf = (lane >> 4) << 4;             // 0 or 16
    const __bf16* pl = src + row * stride + khalf;
    const __bf16* fb = frags + (size_t)lane * 16;
    for (int kc = 0; kc < kchunks; ++kc) {
        ABfrag a;
        a.p.lo = *(const v8bf*)(pl + kc * 32);        // ds_load_b128
        a.p.hi = *(const v8bf*)(pl + kc * 32 + 8);    // ds_load_b128
#pragma unroll
        for (int j = 0; j < TPW; ++j) {
            v16bf b = *(const v16bf*)(fb + (size_t)(kc * NTILES + ntbase + j) * 512);
            acc[j] = __builtin_amdgcn_wmma_f32_16x16x32_bf16(
                         false, a.v, false, b, (short)0, acc[j], false, false);
        }
    }
}

// -------------------------------------------------------------------------
// Persistent ESN recurrence: one workgroup owns 16 batch rows for all
// 2048 steps.  h lives in LDS as bf16; weights stream from L2 as packed
// bf16 fragments.  Last 20 hidden states are spilled to feats (f32).
// -------------------------------------------------------------------------
__global__ __launch_bounds__(WAVES * 32)
void esn_recurrence_kernel(const float* __restrict__ x,
                           const __bf16* __restrict__ fr_in,
                           const __bf16* __restrict__ fr_r,
                           float* __restrict__ feats) {
    __shared__ __attribute__((aligned(16))) __bf16 h_bf[MB * HSTRIDE];
    __shared__ __attribute__((aligned(16))) __bf16 x_bf[MB * XSTRIDE];

    const int tid    = threadIdx.x;
    const int lane   = tid & 31;
    const int wave   = tid >> 5;
    const int b0     = blockIdx.x * MB;
    const int lo     = lane & 15;
    const int hi     = lane >> 4;
    const int ntbase = wave * TPW;

    for (int t = 0; t < TSTEPS; ++t) {
        // stage x_t block [MB, IN_DIM] -> bf16 LDS (coalesced)
        for (int i = tid; i < MB * IN_DIM; i += WAVES * 32) {
            int r = i >> 7;                 // / IN_DIM
            int d = i & (IN_DIM - 1);
            x_bf[r * XSTRIDE + d] =
                (__bf16)x[((size_t)(b0 + r) * TSTEPS + t) * IN_DIM + d];
        }
        __syncthreads();                    // barrier A: h/x updates visible

        v8f acc[TPW];
#pragma unroll
        for (int j = 0; j < TPW; ++j)
#pragma unroll
            for (int e = 0; e < 8; ++e) acc[j][e] = 0.0f;

        gemm_phase(x_bf, XSTRIDE, IN_DIM / 32, fr_in, ntbase, lane, acc);
        if (t > 0)
            gemm_phase(h_bf, HSTRIDE, HID / 32, fr_r, ntbase, lane, acc);

        __syncthreads();                    // barrier B: all h_bf reads done

        // C/D layout (ISA 7.12.2): elem e -> row = e + 8*(lane>=16),
        // col = tile*16 + (lane&15)
#pragma unroll
        for (int j = 0; j < TPW; ++j) {
            int col = wave * (TPW * 16) + j * 16 + lo;
#pragma unroll
            for (int e = 0; e < 8; ++e) {
                int row = e + hi * 8;
                float pre = acc[j][e];
                float hn;
                if (t == 0) {
                    hn = fast_tanh(pre);                 // t==0: no blend
                } else {
                    float hold = (float)h_bf[row * HSTRIDE + col];
                    hn = 0.1f * hold + 0.9f * fast_tanh(pre);
                }
                h_bf[row * HSTRIDE + col] = (__bf16)hn;
                if (t >= TSTEPS - LASTN)
                    feats[((size_t)(b0 + row) * LASTN + (t - (TSTEPS - LASTN))) * HID
                          + col] = hn;
            }
        }
    }
}

// -------------------------------------------------------------------------
// Head: out[256,32] = feats[256,20480] @ lin_wT + lin_b.  One wave per
// 8 batch rows; lin_wT reads coalesced, feats reads are scalar-uniform.
// -------------------------------------------------------------------------
__global__ __launch_bounds__(32)
void head_kernel(const float* __restrict__ feats,
                 const float* __restrict__ lin_wT,
                 const float* __restrict__ lin_b,
                 float* __restrict__ out) {
    const int lane = threadIdx.x;           // output channel o
    const int b0   = blockIdx.x * 8;
    float acc[8];
#pragma unroll
    for (int r = 0; r < 8; ++r) acc[r] = 0.0f;
    const int KF = HID * LASTN;             // 20480
    for (int k = 0; k < KF; ++k) {
        float wv = lin_wT[(size_t)k * 32 + lane];
#pragma unroll
        for (int r = 0; r < 8; ++r)
            acc[r] = fmaf(feats[(size_t)(b0 + r) * KF + k], wv, acc[r]);
    }
#pragma unroll
    for (int r = 0; r < 8; ++r)
        out[(size_t)(b0 + r) * OUT_DIM + lane] = acc[r] + lin_b[lane];
}

// -------------------------------------------------------------------------
extern "C" void kernel_launch(void* const* d_in, const int* in_sizes, int n_in,
                              void* d_out, int out_size, void* d_ws, size_t ws_size,
                              hipStream_t stream) {
    const float* x     = (const float*)d_in[0];
    const float* w_in  = (const float*)d_in[1];
    const float* w_r   = (const float*)d_in[2];
    const float* lin_w = (const float*)d_in[3];
    const float* lin_b = (const float*)d_in[4];
    float* out = (float*)d_out;

    char* ws = (char*)d_ws;
    // workspace layout (all offsets 256B-aligned)
    const size_t sz_frin  = (size_t)IN_DIM * HID * 2;          // 256 KiB
    const size_t sz_frr   = (size_t)HID * HID * 2;             // 2 MiB
    const size_t sz_linwT = (size_t)HID * LASTN * OUT_DIM * 4; // 2.5 MiB
    __bf16* fr_in  = (__bf16*)ws;
    __bf16* fr_r   = (__bf16*)(ws + sz_frin);
    float*  lin_wT = (float*)(ws + sz_frin + sz_frr);
    float*  feats  = (float*)(ws + sz_frin + sz_frr + sz_linwT); // 20 MiB

    // 1) pack weights into WMMA bf16 B-fragments (one-time, L2 resident after)
    {
        int tot_in = (IN_DIM / 32) * (HID / 16) * 32;   // 8192
        pack_frags_kernel<<<(tot_in + 255) / 256, 256, 0, stream>>>(w_in, fr_in, IN_DIM, HID);
        int tot_r = (HID / 32) * (HID / 16) * 32;       // 65536
        pack_frags_kernel<<<(tot_r + 255) / 256, 256, 0, stream>>>(w_r, fr_r, HID, HID);
        int tot_t = OUT_DIM * HID * LASTN;              // 655360
        transpose_linw_kernel<<<(tot_t + 255) / 256, 256, 0, stream>>>(lin_w, lin_wT);
    }

    // 2) persistent recurrence: 16 workgroups x 8 waves, 2048 steps each
    esn_recurrence_kernel<<<BATCH / MB, WAVES * 32, 0, stream>>>(x, fr_in, fr_r, feats);

    // 3) linear head
    head_kernel<<<BATCH / 8, 32, 0, stream>>>(feats, lin_wT, lin_b, out);
}
